// PhyloDisentanglerConv_702
// MI455X (gfx1250) — matrix operand order repaired and checked by reading, not compiled, not run on hardware
//
#include <hip/hip_runtime.h>
#include <hip/hip_bf16.h>

// ---------------------------------------------------------------------------
// PhyloDisentangler VQ block for MI455X (gfx1250, wave32, WMMA).
// All GEMMs (6 conv1x1 + 2 codebook distance searches) run on
// v_wmma_f32_16x16x32_f16. Activations staged f16 pixel-major [P][C] so every
// WMMA fragment is two contiguous 16B loads per lane; weights in native
// [out][in] layout are already the Bt[n][k] fragment layout.
// ---------------------------------------------------------------------------

typedef __attribute__((ext_vector_type(16))) _Float16 v16h;
typedef __attribute__((ext_vector_type(8)))  _Float16 v8h;
typedef __attribute__((ext_vector_type(8)))  float    v8f;

#define P_TOT 4096   // B*RES*RES pixels

__device__ __forceinline__ float silu_f(float x) {
  return x / (1.0f + __expf(-x));
}

// A fragment: 16x32 f16, row-major rows of length lda.
// lane l holds m=l&15, K = (l>>4)*8 + {0..7} and 16 + (l>>4)*8 + {0..7}.
__device__ __forceinline__ v16h load_a_frag(const _Float16* A, int lda) {
  int lane = threadIdx.x & 31;
  const _Float16* p = A + (lane & 15) * lda + ((lane >> 4) << 3);
  v8h lo = *(const v8h*)(p);
  v8h hi = *(const v8h*)(p + 16);
  return __builtin_shufflevector(lo, hi, 0,1,2,3,4,5,6,7,8,9,10,11,12,13,14,15);
}

// B fragment: 32x16 f16 given as Bt[n][k] row-major (weights [out][in]).
// lane l holds n=l&15, K = (l>>4)*8 + {0..7} and 16 + (l>>4)*8 + {0..7}.
__device__ __forceinline__ v16h load_b_frag(const _Float16* Bt, int ldb) {
  int lane = threadIdx.x & 31;
  const _Float16* p = Bt + (lane & 15) * ldb + ((lane >> 4) << 3);
  v8h lo = *(const v8h*)(p);
  v8h hi = *(const v8h*)(p + 16);
  return __builtin_shufflevector(lo, hi, 0,1,2,3,4,5,6,7,8,9,10,11,12,13,14,15);
}

__device__ __forceinline__ v8f wmma_k_loop(const _Float16* A, int lda,
                                           const _Float16* Bt, int ldb,
                                           int K, v8f acc) {
  for (int k0 = 0; k0 < K; k0 += 32) {
    v16h a = load_a_frag(A + k0, lda);
    v16h b = load_b_frag(Bt + k0, ldb);
    acc = __builtin_amdgcn_wmma_f32_16x16x32_f16(false, a, false, b,
                                                 (short)0, acc, false, false);
  }
  return acc;
}

// ---------------------------------------------------------------------------
// prep / convert kernels
// ---------------------------------------------------------------------------

__global__ void cvt_f16_kernel(const float* __restrict__ src,
                               _Float16* __restrict__ dst, int n) {
  int tid = blockIdx.x * 256 + threadIdx.x;
  if (tid < n) dst[tid] = (_Float16)src[tid];
}

__global__ void cbnorm_kernel(const float* __restrict__ cb,
                              float* __restrict__ cbn) {
  int i = blockIdx.x * 256 + threadIdx.x;
  if (i < 1024) {
    float s = 0.0f;
    for (int e = 0; e < 32; ++e) { float v = cb[i * 32 + e]; s += v * v; }
    cbn[i] = -0.5f * s;  // fold -0.5|c|^2 into WMMA C operand
  }
}

// Xs[p][c] = f16(silu(x[b][c][hw])), p = b*256 + hw
__global__ void prep_x_kernel(const float* __restrict__ x,
                              _Float16* __restrict__ Xs) {
  int tid = blockIdx.x * 256 + threadIdx.x;      // p*256 + c, 1048576 total
  int p = tid >> 8, c = tid & 255;
  float v = x[((p >> 8) << 16) + (c << 8) + (p & 255)];
  Xs[tid] = (_Float16)silu_f(v);
}

__global__ void zero2_kernel(float* a) {
  if (threadIdx.x < 2) a[threadIdx.x] = 0.0f;
}

// ---------------------------------------------------------------------------
// GEMM kernels: 8 waves/block, one 16x16 tile per wave, full K loop per wave.
// ---------------------------------------------------------------------------

// Y_f32[p][n] = A @ Bt^T + bias    (conv_in -> H, kept f32 for exact LN stats)
__global__ __launch_bounds__(256)
void gemm_f32out_kernel(const _Float16* __restrict__ A,
                        const _Float16* __restrict__ Bt,
                        const float* __restrict__ bias,
                        float* __restrict__ Y, int N, int K) {
  int wid = threadIdx.x >> 5, lane = threadIdx.x & 31;
  int tile = blockIdx.x * 8 + wid;
  int tn = tile % (N >> 4), tm = tile / (N >> 4);
  v8f acc = {};
  acc = wmma_k_loop(A + tm * 16 * K, K, Bt + tn * 16 * K, K, K, acc);
  int col = tn * 16 + (lane & 15);
  float bv = bias[col];
  int half = lane >> 4;
  for (int r = 0; r < 8; ++r) {
    int p = tm * 16 + r + half * 8;
    Y[p * N + col] = acc[r] + bv;
  }
}

// z = silu(A@Bt^T + bias), scattered to quantize-vector layout:
// row R = p*32 + (j&31), col e = j>>5. Dual f16 (for WMMA) + f32 (for loss).
__global__ __launch_bounds__(256)
void gemm_v_kernel(const _Float16* __restrict__ A,
                   const _Float16* __restrict__ Bt,
                   const float* __restrict__ bias,
                   _Float16* __restrict__ Vh, float* __restrict__ Vf, int K) {
  const int N = 1024;
  int wid = threadIdx.x >> 5, lane = threadIdx.x & 31;
  int tile = blockIdx.x * 8 + wid;
  int tn = tile % (N >> 4), tm = tile / (N >> 4);
  v8f acc = {};
  acc = wmma_k_loop(A + tm * 16 * K, K, Bt + tn * 16 * K, K, K, acc);
  int j = tn * 16 + (lane & 15);
  float bv = bias[j];
  int e = j >> 5, t = j & 31;
  int half = lane >> 4;
  for (int r = 0; r < 8; ++r) {
    int p = tm * 16 + r + half * 8;
    float z = silu_f(acc[r] + bv);
    int R = (p << 5) + t;
    Vh[R * 32 + e] = (_Float16)z;
    Vf[R * 32 + e] = z;
  }
}

// Scat[p][colOff+n] = f16(silu(silu(A@Bt^T + bias)))  (wp_out / wn_out paths)
__global__ __launch_bounds__(256)
void gemm_scat_kernel(const _Float16* __restrict__ A,
                      const _Float16* __restrict__ Bt,
                      const float* __restrict__ bias,
                      _Float16* __restrict__ Scat, int colOff, int N, int K) {
  int wid = threadIdx.x >> 5, lane = threadIdx.x & 31;
  int tile = blockIdx.x * 8 + wid;
  int tn = tile % (N >> 4), tm = tile / (N >> 4);
  v8f acc = {};
  acc = wmma_k_loop(A + tm * 16 * K, K, Bt + tn * 16 * K, K, K, acc);
  int col = tn * 16 + (lane & 15);
  float bv = bias[col];
  int half = lane >> 4;
  for (int r = 0; r < 8; ++r) {
    int p = tm * 16 + r + half * 8;
    Scat[p * 256 + colOff + col] = (_Float16)silu_f(silu_f(acc[r] + bv));
  }
}

// final conv_out: f32 NCHW store to d_out
__global__ __launch_bounds__(256)
void gemm_out_kernel(const _Float16* __restrict__ A,
                     const _Float16* __restrict__ Bt,
                     const float* __restrict__ bias,
                     float* __restrict__ Out) {
  const int N = 256, K = 256;
  int wid = threadIdx.x >> 5, lane = threadIdx.x & 31;
  int tile = blockIdx.x * 8 + wid;
  int tn = tile % (N >> 4), tm = tile / (N >> 4);
  v8f acc = {};
  acc = wmma_k_loop(A + tm * 16 * K, K, Bt + tn * 16 * K, K, K, acc);
  int o = tn * 16 + (lane & 15);
  float bv = bias[o];
  int half = lane >> 4;
  for (int r = 0; r < 8; ++r) {
    int p = tm * 16 + r + half * 8;
    Out[((p >> 8) << 16) + (o << 8) + (p & 255)] = acc[r] + bv;
  }
}

// ---------------------------------------------------------------------------
// VQ: per 16-vector tile, 64 WMMAs sweep the 1024-entry codebook computing
// score = v.c - 0.5|c|^2 ; running argmax + 16-lane xor-butterfly reduction.
// Writes gathered codes Q[p][e*32+t] (f16, feeds next GEMM), optional zq_p
// output, and accumulates sum((zq - z)^2) for the loss.
// ---------------------------------------------------------------------------
__global__ __launch_bounds__(256)
void quantize_kernel(const _Float16* __restrict__ Vh,
                     const float* __restrict__ Vf,
                     const _Float16* __restrict__ cbh,
                     const float* __restrict__ cbf,
                     const float* __restrict__ cbn,
                     _Float16* __restrict__ Q,
                     float* __restrict__ zq_out,   // nullptr for n-path
                     float* __restrict__ loss_acc, int isP) {
  int wid = threadIdx.x >> 5, lane = threadIdx.x & 31;
  int rt = blockIdx.x * 8 + wid;                 // 8192 row tiles
  int ncol = lane & 15, half = lane >> 4;

  v16h a = load_a_frag(Vh + rt * 16 * 32, 32);

  float best[8]; int bidx[8];
  for (int r = 0; r < 8; ++r) { best[r] = -3.0e38f; bidx[r] = 0; }

  for (int ct = 0; ct < 64; ++ct) {
    v16h b = load_b_frag(cbh + ct * 16 * 32, 32);
    int code = ct * 16 + ncol;
    float ci = cbn[code];
    v8f c;
    for (int r = 0; r < 8; ++r) c[r] = ci;
    v8f d = __builtin_amdgcn_wmma_f32_16x16x32_f16(false, a, false, b,
                                                   (short)0, c, false, false);
    for (int r = 0; r < 8; ++r)
      if (d[r] > best[r]) { best[r] = d[r]; bidx[r] = code; }
  }

  float lsum = 0.0f;
  for (int r = 0; r < 8; ++r) {
    float s = best[r]; int i = bidx[r];
    for (int mask = 1; mask < 16; mask <<= 1) {   // stays within 16-lane half
      float os = __shfl_xor(s, mask);
      int   oi = __shfl_xor(i, mask);
      if (os > s || (os == s && oi < i)) { s = os; i = oi; }
    }
    int m = r + half * 8;
    int R = rt * 16 + m;
    int p = R >> 5, t = R & 31;
    float cv0 = cbf[i * 32 + ncol];
    float cv1 = cbf[i * 32 + ncol + 16];
    // reshape_out gather: channel j' = e*32 + t (valid for L=4 and L=1)
    Q[p * 1024 + ncol * 32 + t]        = (_Float16)cv0;
    Q[p * 1024 + (ncol + 16) * 32 + t] = (_Float16)cv1;
    float d0 = cv0 - Vf[R * 32 + ncol];
    float d1 = cv1 - Vf[R * 32 + ncol + 16];
    lsum += d0 * d0 + d1 * d1;
    if (isP) {                                    // zq_p output (B,E,2048,4)
      int bb = p >> 8, hw = p & 255;
      int l = t >> 3, kp = t & 7;
      int n = kp * 256 + hw;
      zq_out[((((bb * 32 + ncol) * 2048) + n) << 2) + l]        = cv0;
      zq_out[((((bb * 32 + ncol + 16) * 2048) + n) << 2) + l]   = cv1;
    }
  }
  for (int mask = 1; mask < 32; mask <<= 1) lsum += __shfl_xor(lsum, mask);
  if (lane == 0) atomicAdd(loss_acc, lsum);
}

// ---------------------------------------------------------------------------
// LayerNorm over (C=128, HW=256) per batch on silu(H[:, c0:c0+128])
// ---------------------------------------------------------------------------
__global__ void ln_reduce_kernel(const float* __restrict__ H, int c0,
                                 float* __restrict__ sums) {
  __shared__ float sh[512];
  int b = blockIdx.x;
  float s = 0.0f, s2 = 0.0f;
  for (int i = threadIdx.x; i < 32768; i += 256) {
    int cl = i >> 8, hw = i & 255;
    float v = silu_f(H[((((b << 8) + hw)) << 8) + c0 + cl]);
    s += v; s2 += v * v;
  }
  sh[threadIdx.x] = s; sh[256 + threadIdx.x] = s2;
  __syncthreads();
  for (int st = 128; st > 0; st >>= 1) {
    if (threadIdx.x < (unsigned)st) {
      sh[threadIdx.x] += sh[threadIdx.x + st];
      sh[256 + threadIdx.x] += sh[256 + threadIdx.x + st];
    }
    __syncthreads();
  }
  if (threadIdx.x == 0) { sums[b * 2] = sh[0]; sums[b * 2 + 1] = sh[256]; }
}

__global__ void ln_apply_kernel(const float* __restrict__ H,
                                const float* __restrict__ sums,
                                const float* __restrict__ lnw,
                                const float* __restrict__ lnb,
                                int c0, _Float16* __restrict__ Aout) {
  int tid = blockIdx.x * 256 + threadIdx.x;      // 16 * 32768
  int b = tid >> 15, i = tid & 32767;
  int cl = i >> 8, hw = i & 255;
  float mu  = sums[b * 2] * (1.0f / 32768.0f);
  float var = sums[b * 2 + 1] * (1.0f / 32768.0f) - mu * mu;
  float rs  = rsqrtf(var + 1e-5f);
  float v = silu_f(H[((((b << 8) + hw)) << 8) + c0 + cl]);
  Aout[(((b << 8) + hw) << 7) + cl] =
      (_Float16)((v - mu) * rs * lnw[(cl << 8) + hw] + lnb[(cl << 8) + hw]);
}

__global__ void loss_final_kernel(const float* __restrict__ acc,
                                  float* __restrict__ out) {
  if (threadIdx.x == 0)
    out[0] = 1.25f * (acc[0] + acc[1]) * (1.0f / 4194304.0f);
}

// ---------------------------------------------------------------------------
extern "C" void kernel_launch(void* const* d_in, const int* in_sizes, int n_in,
                              void* d_out, int out_size, void* d_ws, size_t ws_size,
                              hipStream_t stream) {
  (void)in_sizes; (void)n_in; (void)out_size; (void)ws_size;

  const float* x        = (const float*)d_in[0];
  const float* w_in     = (const float*)d_in[1];
  const float* b_in     = (const float*)d_in[2];
  const float* lnp_w    = (const float*)d_in[3];
  const float* lnp_b    = (const float*)d_in[4];
  const float* wp_in    = (const float*)d_in[5];
  const float* bp_in    = (const float*)d_in[6];
  const float* wp_out   = (const float*)d_in[7];
  const float* bp_out   = (const float*)d_in[8];
  const float* lnn_w    = (const float*)d_in[9];
  const float* lnn_b    = (const float*)d_in[10];
  const float* wn_in    = (const float*)d_in[11];
  const float* bn_in    = (const float*)d_in[12];
  const float* wn_out   = (const float*)d_in[13];
  const float* bn_out   = (const float*)d_in[14];
  const float* codebook = (const float*)d_in[15];
  const float* wout     = (const float*)d_in[16];
  const float* b_out    = (const float*)d_in[17];

  float* out      = (float*)d_out;
  float* loss_out = out + 1048576;
  float* zq_out   = out + 1048577;

  char* ws = (char*)d_ws;
  size_t off = 0;
  auto alloc = [&](size_t bytes) -> void* {
    void* p = ws + off;
    off = (off + bytes + 255) & ~(size_t)255;
    return p;
  };

  _Float16* Xs      = (_Float16*)alloc((size_t)P_TOT * 256 * 2);
  _Float16* win_h   = (_Float16*)alloc(65536 * 2);
  _Float16* wpin_h  = (_Float16*)alloc(131072 * 2);
  _Float16* wpout_h = (_Float16*)alloc(131072 * 2);
  _Float16* wnin_h  = (_Float16*)alloc(131072 * 2);
  _Float16* wnout_h = (_Float16*)alloc(131072 * 2);
  _Float16* wout_h  = (_Float16*)alloc(65536 * 2);
  _Float16* cb_h    = (_Float16*)alloc(32768 * 2);
  float*    cb_n    = (float*)alloc(1024 * 4);
  float*    H       = (float*)alloc((size_t)P_TOT * 256 * 4);
  float*    lnsum_p = (float*)alloc(128);
  float*    lnsum_n = (float*)alloc(128);
  _Float16* Ap      = (_Float16*)alloc((size_t)P_TOT * 128 * 2);
  _Float16* An      = (_Float16*)alloc((size_t)P_TOT * 128 * 2);
  _Float16* Vph     = (_Float16*)alloc((size_t)131072 * 32 * 2);
  float*    Vpf     = (float*)alloc((size_t)131072 * 32 * 4);
  _Float16* Vnh     = (_Float16*)alloc((size_t)131072 * 32 * 2);
  float*    Vnf     = (float*)alloc((size_t)131072 * 32 * 4);
  _Float16* Qp      = (_Float16*)alloc((size_t)P_TOT * 1024 * 2);
  _Float16* Qn      = (_Float16*)alloc((size_t)P_TOT * 1024 * 2);
  _Float16* Scat    = (_Float16*)alloc((size_t)P_TOT * 256 * 2);
  float*    lossacc = (float*)alloc(256);

  zero2_kernel<<<1, 32, 0, stream>>>(lossacc);

  cvt_f16_kernel<<<256, 256, 0, stream>>>(w_in,   win_h,   65536);
  cvt_f16_kernel<<<512, 256, 0, stream>>>(wp_in,  wpin_h,  131072);
  cvt_f16_kernel<<<512, 256, 0, stream>>>(wp_out, wpout_h, 131072);
  cvt_f16_kernel<<<512, 256, 0, stream>>>(wn_in,  wnin_h,  131072);
  cvt_f16_kernel<<<512, 256, 0, stream>>>(wn_out, wnout_h, 131072);
  cvt_f16_kernel<<<256, 256, 0, stream>>>(wout,   wout_h,  65536);
  cvt_f16_kernel<<<128, 256, 0, stream>>>(codebook, cb_h,  32768);
  cbnorm_kernel<<<4, 256, 0, stream>>>(codebook, cb_n);

  prep_x_kernel<<<4096, 256, 0, stream>>>(x, Xs);

  // conv_in: [4096x256] = Xs[4096x256] @ win^T  -> f32 H
  gemm_f32out_kernel<<<512, 256, 0, stream>>>(Xs, win_h, b_in, H, 256, 256);

  ln_reduce_kernel<<<16, 256, 0, stream>>>(H, 0, lnsum_p);
  ln_apply_kernel<<<2048, 256, 0, stream>>>(H, lnsum_p, lnp_w, lnp_b, 0, Ap);
  ln_reduce_kernel<<<16, 256, 0, stream>>>(H, 128, lnsum_n);
  ln_apply_kernel<<<2048, 256, 0, stream>>>(H, lnsum_n, lnn_w, lnn_b, 128, An);

  // wp_in / wn_in: [4096x1024], K=128, fused silu + vector-layout scatter
  gemm_v_kernel<<<2048, 256, 0, stream>>>(Ap, wpin_h, bp_in, Vph, Vpf, 128);
  gemm_v_kernel<<<2048, 256, 0, stream>>>(An, wnin_h, bn_in, Vnh, Vnf, 128);

  // VQ: 131072 vectors x 1024 codes, K=32 (one WMMA per code tile)
  quantize_kernel<<<1024, 256, 0, stream>>>(Vph, Vpf, cb_h, codebook, cb_n,
                                            Qp, zq_out, lossacc + 0, 1);
  quantize_kernel<<<1024, 256, 0, stream>>>(Vnh, Vnf, cb_h, codebook, cb_n,
                                            Qn, nullptr, lossacc + 1, 0);

  // wp_out / wn_out: [4096x128], K=1024, fused silu(silu(.)) into Scat
  gemm_scat_kernel<<<256, 256, 0, stream>>>(Qp, wpout_h, bp_out, Scat, 0, 128, 1024);
  gemm_scat_kernel<<<256, 256, 0, stream>>>(Qn, wnout_h, bn_out, Scat, 128, 128, 1024);

  // conv_out: [4096x256], K=256 -> f32 NCHW d_out
  gemm_out_kernel<<<512, 256, 0, stream>>>(Scat, wout_h, b_out, out);

  loss_final_kernel<<<1, 32, 0, stream>>>(lossacc, loss_out);
}
